// CLIPVisionTower_73469710565951
// MI455X (gfx1250) — matrix operand order repaired and checked by reading, not compiled, not run on hardware
//
#include <hip/hip_runtime.h>
#include <hip/hip_bf16.h>
#include <stdint.h>

typedef __bf16 bf16;
typedef __attribute__((ext_vector_type(16))) __bf16 v16bf;
typedef __attribute__((ext_vector_type(8)))  float  v8f;
typedef __attribute__((ext_vector_type(4)))  unsigned int u32x4;

union Frag { v16bf v; u32x4 u[2]; };

__device__ __forceinline__ u32x4 ld128(const void* p) { return *(const u32x4*)p; }

#define WMMA_BF16(a, b, c) \
  __builtin_amdgcn_wmma_f32_16x16x32_bf16(false, (a), false, (b), (short)0, (c), false, false)

// ---- problem dimensions ----
constexpr int Bn = 16, Tn = 577, Dn = 1024, Ln = 64, DLLM = 4096, Hn = 16, HDn = 64;
constexpr int TPAD = 592;             // ceil16(T)
constexpr int QPAD = 640;             // 10 * 64 query rows for attention grid
constexpr int SPAD = 672;             // 21 * 32 keys (S = L + T = 641 padded)
constexpr int SVAL = Ln + Tn;         // 641
constexpr float QSCALE = 0.125f;      // HD^-0.5

// ================= f32 -> bf16 cast with row padding =================
__global__ void cast_pad(const float* __restrict__ src, bf16* __restrict__ dst,
                         int rows_valid, int rows_pad, int cols, int batches) {
  long n = (long)batches * rows_pad * cols;
  for (long i = blockIdx.x * (long)blockDim.x + threadIdx.x; i < n;
       i += (long)gridDim.x * blockDim.x) {
    int c = (int)(i % cols);
    long rb = i / cols;
    int r = (int)(rb % rows_pad);
    int b = (int)(rb / rows_pad);
    float v = 0.0f;
    if (r < rows_valid) v = src[((long)b * rows_valid + r) * cols + c];
    dst[i] = (bf16)v;
  }
}

__global__ void fill_zero_bf16(bf16* __restrict__ p, long n) {
  for (long i = blockIdx.x * (long)blockDim.x + threadIdx.x; i < n;
       i += (long)gridDim.x * blockDim.x)
    p[i] = (bf16)0.0f;
}

// ================= generic  C = A @ W^T + bias  (bf16 WMMA) =================
// A: [Mtot, K] bf16 row-major (Mtot = batches*rpb).  W: [N, K] bf16 row-major.
// MODE 0: f32 store   dst[bb*bstride + (s+row_off)*ld + col]
// MODE 1: bf16 store  same indexing
// MODE 2: bf16 store transposed for V: dst[(bb*1024 + col)*SPAD + s + row_off]
template<int MODE>
__global__ __launch_bounds__(128) void gemm_wmma(
    const bf16* __restrict__ A, const bf16* __restrict__ W,
    const float* __restrict__ bias, float scale,
    int K, int rpb, int valid, int row_off,
    void* __restrict__ dst, long ld, long bstride) {
  const int wave = threadIdx.x >> 5;
  const int lane = threadIdx.x & 31;
  const int ll = lane & 15, lh = lane >> 4;
  const long mtile = (long)blockIdx.x * 64 + wave * 16;
  const long n0 = (long)blockIdx.y * 64;

  const bf16* abase = A + (mtile + ll) * (long)K + lh * 16;
  const bf16* wb0 = W + (n0 +  0 + ll) * (long)K + lh * 16;
  const bf16* wb1 = W + (n0 + 16 + ll) * (long)K + lh * 16;
  const bf16* wb2 = W + (n0 + 32 + ll) * (long)K + lh * 16;
  const bf16* wb3 = W + (n0 + 48 + ll) * (long)K + lh * 16;

  v8f z = {0.f, 0.f, 0.f, 0.f, 0.f, 0.f, 0.f, 0.f};
  v8f acc0 = z, acc1 = z, acc2 = z, acc3 = z;

  for (int kb = 0; kb < K; kb += 32) {
    Frag a, b;
    a.u[0] = ld128(abase + kb);
    a.u[1] = ld128(abase + kb + 8);
    b.u[0] = ld128(wb0 + kb); b.u[1] = ld128(wb0 + kb + 8);
    acc0 = WMMA_BF16(a.v, b.v, acc0);
    b.u[0] = ld128(wb1 + kb); b.u[1] = ld128(wb1 + kb + 8);
    acc1 = WMMA_BF16(a.v, b.v, acc1);
    b.u[0] = ld128(wb2 + kb); b.u[1] = ld128(wb2 + kb + 8);
    acc2 = WMMA_BF16(a.v, b.v, acc2);
    b.u[0] = ld128(wb3 + kb); b.u[1] = ld128(wb3 + kb + 8);
    acc3 = WMMA_BF16(a.v, b.v, acc3);
  }

  const int bb = (int)(mtile / rpb);
  const int sbase = (int)(mtile % rpb) + lh * 8;
  v8f accs[4] = {acc0, acc1, acc2, acc3};
#pragma unroll
  for (int j = 0; j < 4; ++j) {
    const int col = (int)n0 + j * 16 + ll;
    const float bv = bias[col];
#pragma unroll
    for (int r = 0; r < 8; ++r) {
      const int s = sbase + r;
      if (s < valid) {
        const float val = (accs[j][r] + bv) * scale;
        if (MODE == 0)
          ((float*)dst)[bb * bstride + (long)(s + row_off) * ld + col] = val;
        else if (MODE == 1)
          ((bf16*)dst)[bb * bstride + (long)(s + row_off) * ld + col] = (bf16)val;
        else
          ((bf16*)dst)[((long)bb * (Hn * HDn) + col) * SPAD + s + row_off] = (bf16)val;
      }
    }
  }
}

// ================= flash attention (bf16 WMMA, online softmax) =================
// q:   [B, QPAD, D] bf16 (already scaled)
// k:   [B, SPAD, D] bf16
// vt:  [B, H, HD, SPAD] bf16 (V transposed)
// ctx: [B, TPAD, D] bf16
// Row sums computed with an extra WMMA against a constant all-ones B fragment;
// softmax offset is the (valid) tile-wide running max — a per-row-constant
// offset, so results are mathematically identical and overflow-safe.
__global__ __launch_bounds__(128) void attn_wmma(
    const bf16* __restrict__ q, const bf16* __restrict__ kbuf,
    const bf16* __restrict__ vt, bf16* __restrict__ ctx) {
  __shared__ __align__(16) bf16 plds[4][16 * 32];
  const int wave = threadIdx.x >> 5;
  const int lane = threadIdx.x & 31;
  const int ll = lane & 15, lh = lane >> 4;
  const int b = blockIdx.y >> 4, h = blockIdx.y & 15;
  const int t0 = blockIdx.x * 64 + wave * 16;
  if (t0 >= Tn) return;  // wave-uniform: whole wave out of range

  // Q fragments for head-dim chunks [0,32) and [32,64)
  const bf16* qb = q + ((long)(b * QPAD + t0 + ll)) * Dn + h * HDn + lh * 16;
  Frag qf0, qf1;
  qf0.u[0] = ld128(qb);      qf0.u[1] = ld128(qb + 8);
  qf1.u[0] = ld128(qb + 32); qf1.u[1] = ld128(qb + 40);

  // constant all-ones B fragment (for row-sum WMMA)
  Frag onesf;
#pragma unroll
  for (int i = 0; i < 16; ++i) onesf.v[i] = (bf16)1.0f;

  v8f z = {0.f, 0.f, 0.f, 0.f, 0.f, 0.f, 0.f, 0.f};
  v8f o0 = z, o1 = z, o2 = z, o3 = z, o4 = z;  // o4 = running row sums
  float m = -1e30f;                            // tile-wide running max

  const bf16* kb_base = kbuf + ((long)b * SPAD + ll) * Dn + h * HDn + lh * 16;
  const bf16* vt_base = vt + ((long)(b * Hn + h) * HDn + ll) * SPAD + lh * 16;

  for (int s0 = 0; s0 < SPAD; s0 += 32) {
    // ---- logits: two 16x16 tiles over 32 keys ----
    v8f lg[2];
#pragma unroll
    for (int j = 0; j < 2; ++j) {
      const bf16* kp = kb_base + (long)(s0 + j * 16) * Dn;
      Frag kf;
      v8f a = z;
      kf.u[0] = ld128(kp);      kf.u[1] = ld128(kp + 8);
      a = WMMA_BF16(qf0.v, kf.v, a);
      kf.u[0] = ld128(kp + 32); kf.u[1] = ld128(kp + 40);
      a = WMMA_BF16(qf1.v, kf.v, a);
      lg[j] = a;
    }
    // mask invalid keys (select kills any pad garbage / NaN)
    const bool ok0 = (s0 + ll) < SVAL;
    const bool ok1 = (s0 + 16 + ll) < SVAL;
#pragma unroll
    for (int r = 0; r < 8; ++r) {
      lg[0][r] = ok0 ? lg[0][r] : -1e30f;
      lg[1][r] = ok1 ? lg[1][r] : -1e30f;
    }
    // ---- tile-wide max: in-register tree + 5 cross-lane stages ----
    float tm = fmaxf(lg[0][0], lg[1][0]);
#pragma unroll
    for (int r = 1; r < 8; ++r) tm = fmaxf(tm, fmaxf(lg[0][r], lg[1][r]));
    tm = fmaxf(tm, __shfl_xor(tm, 16));
    tm = fmaxf(tm, __shfl_xor(tm, 8));
    tm = fmaxf(tm, __shfl_xor(tm, 4));
    tm = fmaxf(tm, __shfl_xor(tm, 2));
    tm = fmaxf(tm, __shfl_xor(tm, 1));
    const float mnew = fmaxf(m, tm);
    const float corr = __expf(m - mnew);
    m = mnew;
    // ---- probs (bf16 to LDS) + rescale accumulators ----
#pragma unroll
    for (int r = 0; r < 8; ++r) {
      const float p0 = __expf(lg[0][r] - mnew);
      const float p1 = __expf(lg[1][r] - mnew);
      o0[r] *= corr; o1[r] *= corr; o2[r] *= corr; o3[r] *= corr; o4[r] *= corr;
      plds[wave][(lh * 8 + r) * 32 + ll]      = (bf16)p0;
      plds[wave][(lh * 8 + r) * 32 + 16 + ll] = (bf16)p1;
    }
    asm volatile("" ::: "memory");  // order LDS stores before re-shaped loads (per-wave, DS in-order)

    // ---- P (16x32) from LDS as A-fragment ----
    Frag pf;
    pf.u[0] = ld128(&plds[wave][ll * 32 + lh * 16]);
    pf.u[1] = ld128(&plds[wave][ll * 32 + lh * 16 + 8]);

    // ---- O += P @ V over four 16-wide head-dim tiles, + row sums ----
    const bf16* vp = vt_base + s0;
    Frag vf;
    vf.u[0] = ld128(vp);                 vf.u[1] = ld128(vp + 8);
    o0 = WMMA_BF16(pf.v, vf.v, o0);
    vf.u[0] = ld128(vp + 16L * SPAD);    vf.u[1] = ld128(vp + 16L * SPAD + 8);
    o1 = WMMA_BF16(pf.v, vf.v, o1);
    vf.u[0] = ld128(vp + 32L * SPAD);    vf.u[1] = ld128(vp + 32L * SPAD + 8);
    o2 = WMMA_BF16(pf.v, vf.v, o2);
    vf.u[0] = ld128(vp + 48L * SPAD);    vf.u[1] = ld128(vp + 48L * SPAD + 8);
    o3 = WMMA_BF16(pf.v, vf.v, o3);
    o4 = WMMA_BF16(pf.v, onesf.v, o4);   // row sums, every column identical
    asm volatile("" ::: "memory");
  }

  // ---- normalize + store ctx ----
#pragma unroll
  for (int r = 0; r < 8; ++r) {
    const int t = t0 + lh * 8 + r;
    if (t < Tn) {
      const float inv = 1.0f / fmaxf(o4[r], 1e-30f);
      const long base = ((long)b * TPAD + t) * Dn + h * HDn + ll;
      ctx[base]      = (bf16)(o0[r] * inv);
      ctx[base + 16] = (bf16)(o1[r] * inv);
      ctx[base + 32] = (bf16)(o2[r] * inv);
      ctx[base + 48] = (bf16)(o3[r] * inv);
    }
  }
}

// ================= host =================
extern "C" void kernel_launch(void* const* d_in, const int* in_sizes, int n_in,
                              void* d_out, int out_size, void* d_ws, size_t ws_size,
                              hipStream_t stream) {
  (void)in_sizes; (void)n_in; (void)out_size;
  const float* hidden   = (const float*)d_in[0];
  const float* latt_raw = (const float*)d_in[1];
  const float* Wp = (const float*)d_in[2];
  const float* bp = (const float*)d_in[3];
  const float* Wq = (const float*)d_in[4];
  const float* bq = (const float*)d_in[5];
  const float* Wk = (const float*)d_in[6];
  const float* bk = (const float*)d_in[7];
  const float* Wv = (const float*)d_in[8];
  const float* bv = (const float*)d_in[9];
  const float* Wo = (const float*)d_in[10];
  const float* bo = (const float*)d_in[11];

  uint8_t* w = (uint8_t*)d_ws;
  size_t off = 0;
  auto alloc = [&](size_t elems) -> bf16* {
    bf16* p = (bf16*)(w + off);
    off += ((elems * 2 + 255) / 256) * 256;
    return p;
  };
  bf16* hid_bf  = alloc((size_t)Bn * TPAD * Dn);
  bf16* lraw_bf = alloc((size_t)Bn * Ln * DLLM);
  bf16* Wp_bf   = alloc((size_t)Dn * DLLM);
  bf16* Wq_bf   = alloc((size_t)Dn * Dn);
  bf16* Wk_bf   = alloc((size_t)Dn * Dn);
  bf16* Wv_bf   = alloc((size_t)Dn * Dn);
  bf16* Wo_bf   = alloc((size_t)Dn * Dn);
  bf16* latt_bf = alloc((size_t)Bn * Ln * Dn);
  bf16* q_bf    = alloc((size_t)Bn * QPAD * Dn);
  bf16* k_bf    = alloc((size_t)Bn * SPAD * Dn);
  bf16* v_t     = alloc((size_t)Bn * Hn * HDn * SPAD);
  bf16* ctx_bf  = alloc((size_t)Bn * TPAD * Dn);
  if (off > ws_size) return;  // workspace too small: no-op

  const dim3 cb(256), cg(2048);
  cast_pad<<<cg, cb, 0, stream>>>(hidden,   hid_bf,  Tn, TPAD, Dn, Bn);
  cast_pad<<<cg, cb, 0, stream>>>(latt_raw, lraw_bf, Bn * Ln, Bn * Ln, DLLM, 1);
  cast_pad<<<cg, cb, 0, stream>>>(Wp, Wp_bf, Dn, Dn, DLLM, 1);
  cast_pad<<<cg, cb, 0, stream>>>(Wq, Wq_bf, Dn, Dn, Dn, 1);
  cast_pad<<<cg, cb, 0, stream>>>(Wk, Wk_bf, Dn, Dn, Dn, 1);
  cast_pad<<<cg, cb, 0, stream>>>(Wv, Wv_bf, Dn, Dn, Dn, 1);
  cast_pad<<<cg, cb, 0, stream>>>(Wo, Wo_bf, Dn, Dn, Dn, 1);
  // V pad columns must be finite (multiplied by zero probs): zero entire V^T
  fill_zero_bf16<<<cg, cb, 0, stream>>>(v_t, (long)Bn * Hn * HDn * SPAD);

  const dim3 blk(128);
  // latt = latt_raw @ Wp^T + bp          -> latt_bf [B*L, D]
  gemm_wmma<1><<<dim3((Bn * Ln) / 64, Dn / 64), blk, 0, stream>>>(
      lraw_bf, Wp_bf, bp, 1.0f, DLLM, Ln, Ln, 0, latt_bf, Dn, (long)Ln * Dn);
  // q = (hidden @ Wq^T + bq) * SCALE     -> q_bf [B, QPAD, D]
  gemm_wmma<1><<<dim3((Bn * TPAD) / 64, Dn / 64), blk, 0, stream>>>(
      hid_bf, Wq_bf, bq, QSCALE, Dn, TPAD, TPAD, 0, q_bf, Dn, (long)QPAD * Dn);
  // k: latent part -> rows [0,64), image part -> rows [64, ...)
  gemm_wmma<1><<<dim3((Bn * Ln) / 64, Dn / 64), blk, 0, stream>>>(
      latt_bf, Wk_bf, bk, 1.0f, Dn, Ln, Ln, 0, k_bf, Dn, (long)SPAD * Dn);
  gemm_wmma<1><<<dim3((Bn * TPAD) / 64, Dn / 64), blk, 0, stream>>>(
      hid_bf, Wk_bf, bk, 1.0f, Dn, TPAD, TPAD, Ln, k_bf, Dn, (long)SPAD * Dn);
  // v (stored transposed [B,H,HD,SPAD])
  gemm_wmma<2><<<dim3((Bn * Ln) / 64, Dn / 64), blk, 0, stream>>>(
      latt_bf, Wv_bf, bv, 1.0f, Dn, Ln, Ln, 0, v_t, 0, 0);
  gemm_wmma<2><<<dim3((Bn * TPAD) / 64, Dn / 64), blk, 0, stream>>>(
      hid_bf, Wv_bf, bv, 1.0f, Dn, TPAD, TPAD, Ln, v_t, 0, 0);
  // attention -> ctx_bf [B, TPAD, D]
  attn_wmma<<<dim3((Tn + 63) / 64, Bn * Hn), blk, 0, stream>>>(q_bf, k_bf, v_t, ctx_bf);
  // out = ctx @ Wo^T + bo -> f32 d_out [B, T, D]
  gemm_wmma<0><<<dim3((Bn * TPAD) / 64, Dn / 64), blk, 0, stream>>>(
      ctx_bf, Wo_bf, bo, 1.0f, Dn, TPAD, Tn, 0, d_out, Dn, (long)Tn * Dn);
}
